// TransformerBottleneck_50912542327153
// MI455X (gfx1250) — matrix operand order repaired and checked by reading, not compiled, not run on hardware
//
#include <hip/hip_runtime.h>
#include <hip/hip_bf16.h>
#include <cstdint>
#include <cstddef>

#define DD     512
#define NN     512
#define BB     8
#define HH     8
#define DHD    64
#define INNERD 512
#define CDIMD  768
#define MTOK   77
#define DEPTHN 4

typedef __attribute__((ext_vector_type(16))) __bf16 v16bf;
typedef __attribute__((ext_vector_type(8)))  float  v8f;

union Frag {
    v16bf v;
    uint4 q[2];
    unsigned short u[16];
};

__device__ __forceinline__ unsigned short f2bf(float f) {
    union { float f; unsigned int u; } c; c.f = f;
    unsigned int r = c.u + 0x7FFFu + ((c.u >> 16) & 1u);
    return (unsigned short)(r >> 16);
}

// ---------------- CDNA5 async global->LDS copy (guarded; plain fallback) ----
#if __has_builtin(__builtin_amdgcn_global_load_async_to_lds_b128) && \
    __has_builtin(__builtin_amdgcn_s_wait_asynccnt)
#define ASYNC_LDS 1
typedef int v4i_t __attribute__((vector_size(16)));
typedef __attribute__((address_space(3))) v4i_t* as3_v4i;
__device__ __forceinline__ void cp16_g2l(const void* g, void* l) {
    __builtin_amdgcn_global_load_async_to_lds_b128((v4i_t*)g, (as3_v4i)l, 0, 0);
}
__device__ __forceinline__ void cp_wait() { __builtin_amdgcn_s_wait_asynccnt(0); }
#else
#define ASYNC_LDS 0
__device__ __forceinline__ void cp16_g2l(const void* g, void* l) {
    *(uint4*)l = *(const uint4*)g;
}
__device__ __forceinline__ void cp_wait() {}
#endif

// ---------------------------------------------------------------- utilities
__global__ void cvt_bf16_kernel(const float* __restrict__ s,
                                unsigned short* __restrict__ d, int n) {
    int i = blockIdx.x * 256 + threadIdx.x;
    if (i < n) d[i] = f2bf(s[i]);
}

// [L, K, N] f32 -> [L, N, K] bf16 (weight transpose, done once per call)
__global__ void cvt_wt_kernel(const float* __restrict__ s,
                              unsigned short* __restrict__ d,
                              int K, int N, int total) {
    int idx = blockIdx.x * 256 + threadIdx.x;
    if (idx >= total) return;
    int k = idx % K;
    int n = (idx / K) % N;
    int l = idx / (K * N);
    d[idx] = f2bf(s[(size_t)l * K * N + (size_t)k * N + n]);
}

// x [B, D, N] -> xt [B, N, D]
__global__ void transpose_in_kernel(const float* __restrict__ x,
                                    float* __restrict__ xt) {
    int idx = blockIdx.x * 256 + threadIdx.x;
    int c = idx & 511;
    int n = (idx >> 9) & 511;
    int b = idx >> 18;
    xt[((size_t)b * NN + n) * DD + c] = x[((size_t)b * DD + c) * NN + n];
}

// xt [B, N, D] -> out [B, D, N]
__global__ void transpose_out_kernel(const float* __restrict__ xt,
                                     float* __restrict__ out) {
    int idx = blockIdx.x * 256 + threadIdx.x;
    int n = idx & 511;
    int c = (idx >> 9) & 511;
    int b = idx >> 18;
    out[((size_t)b * DD + c) * NN + n] = xt[((size_t)b * NN + n) * DD + c];
}

// LayerNorm over D=512; one row per block of 128 threads. Writes f32 + bf16.
__global__ __launch_bounds__(128) void ln_kernel(
    const float* __restrict__ x, const float* __restrict__ g,
    const float* __restrict__ bta, float* __restrict__ y,
    unsigned short* __restrict__ ybf) {
    __shared__ float s1[128], s2[128];
    int r = blockIdx.x, t = threadIdx.x;
    const float* xr = x + (size_t)r * DD;
    float4 v = *(const float4*)(xr + t * 4);
    float sum = v.x + v.y + v.z + v.w;
    float sq  = v.x * v.x + v.y * v.y + v.z * v.z + v.w * v.w;
    s1[t] = sum; s2[t] = sq;
    __syncthreads();
    for (int off = 64; off > 0; off >>= 1) {
        if (t < off) { s1[t] += s1[t + off]; s2[t] += s2[t + off]; }
        __syncthreads();
    }
    float mu  = s1[0] * (1.0f / 512.0f);
    float var = s2[0] * (1.0f / 512.0f) - mu * mu;
    float rs  = rsqrtf(var + 1e-5f);
    float4 gg = *(const float4*)(g + t * 4);
    float4 bb = *(const float4*)(bta + t * 4);
    float4 o;
    o.x = (v.x - mu) * rs * gg.x + bb.x;
    o.y = (v.y - mu) * rs * gg.y + bb.y;
    o.z = (v.z - mu) * rs * gg.z + bb.z;
    o.w = (v.w - mu) * rs * gg.w + bb.w;
    *(float4*)(y + (size_t)r * DD + t * 4) = o;
    union { unsigned short h[4]; uint2 u; } pk;
    pk.h[0] = f2bf(o.x); pk.h[1] = f2bf(o.y);
    pk.h[2] = f2bf(o.z); pk.h[3] = f2bf(o.w);
    *(uint2*)(ybf + (size_t)r * DD + t * 4) = pk.u;
}

// ---------------------------------------------------------------- WMMA GEMM
// C[R,N] = A[R,K](bf16, row-major) x Wt[N,K](bf16, pre-transposed), f32 accum.
// Block tile 128x128, 8 waves: wave w -> M-block (w&3)*32, N-block (w>>2)*64.
// Each wave: 2x4 = 8 accumulators, 8 wmma per 32-wide k-step.
// Double-buffered LDS, async global->LDS staging, single barrier per step.
__global__ __launch_bounds__(256) void gemm_bf16_kernel(
    const unsigned short* __restrict__ A, const unsigned short* __restrict__ Wt,
    const float* __restrict__ bias, const float* __restrict__ res,
    float* __restrict__ outf, unsigned short* __restrict__ outb,
    int R, int K, int N, int do_silu) {
    __shared__ unsigned short Al[2][128 * 32];   // [m][k]
    __shared__ unsigned short Bl[2][128 * 32];   // [n][k]
    const int t = threadIdx.x;
    const int lane = t & 31, wid = t >> 5;
    const int r0 = blockIdx.x * 128, n0 = blockIdx.y * 128;
    const int Mw = wid & 3, Nw = wid >> 2;
    const int m = lane & 15, hi = lane >> 4;
    const int smm = t >> 1;            // 0..127
    const int skk = (t & 1) * 16;      // 0 or 16

    v8f zero = {};
    v8f acc[2][4];
    #pragma unroll
    for (int i = 0; i < 2; ++i)
        #pragma unroll
        for (int j = 0; j < 4; ++j) acc[i][j] = zero;

    auto stage = [&](int buf, int k0) {
        // A tile 128x32
        unsigned short* la = &Al[buf][smm * 32 + skk];
        if (r0 + smm < R) {
            const unsigned short* ga = A + (size_t)(r0 + smm) * K + k0 + skk;
            cp16_g2l(ga, la);
            cp16_g2l(ga + 8, la + 8);
        } else {
            uint4 z = make_uint4(0u, 0u, 0u, 0u);
            *(uint4*)la = z;
            *(uint4*)(la + 8) = z;
        }
        // B tile 128x32 (k-contiguous rows of Wt)
        const unsigned short* gb = Wt + (size_t)(n0 + smm) * K + k0 + skk;
        unsigned short* lb = &Bl[buf][smm * 32 + skk];
        cp16_g2l(gb, lb);
        cp16_g2l(gb + 8, lb + 8);
    };

    const int nsteps = K >> 5;
    stage(0, 0);
    cp_wait();
    __syncthreads();

    for (int s = 0; s < nsteps; ++s) {
        const int cur = s & 1;
        if (s + 1 < nsteps) stage(cur ^ 1, (s + 1) * 32);

        const unsigned short* Ab = Al[cur];
        const unsigned short* Bb = Bl[cur];
        Frag af[2];
        #pragma unroll
        for (int ti = 0; ti < 2; ++ti) {
            const unsigned short* ap = Ab + (Mw * 32 + ti * 16 + m) * 32;
            af[ti].q[0] = *(const uint4*)(ap + hi * 16);
            af[ti].q[1] = *(const uint4*)(ap + hi * 16 + 8);
        }
        Frag bfr[4];
        #pragma unroll
        for (int tj = 0; tj < 4; ++tj) {
            const unsigned short* bp = Bb + (Nw * 64 + tj * 16 + m) * 32 + hi * 16;
            bfr[tj].q[0] = *(const uint4*)bp;
            bfr[tj].q[1] = *(const uint4*)(bp + 8);
        }
        #pragma unroll
        for (int ti = 0; ti < 2; ++ti)
            #pragma unroll
            for (int tj = 0; tj < 4; ++tj)
                acc[ti][tj] = __builtin_amdgcn_wmma_f32_16x16x32_bf16(
                    false, af[ti].v, false, bfr[tj].v, (short)0,
                    acc[ti][tj], false, false);

        cp_wait();
        __syncthreads();
    }

    #pragma unroll
    for (int ti = 0; ti < 2; ++ti)
        #pragma unroll
        for (int tj = 0; tj < 4; ++tj) {
            int col = n0 + Nw * 64 + tj * 16 + m;
            #pragma unroll
            for (int j = 0; j < 8; ++j) {
                int row = r0 + Mw * 32 + ti * 16 + hi * 8 + j;
                if (row < R) {
                    float v = acc[ti][tj][j];
                    if (bias) v += bias[col];
                    if (do_silu) v = v / (1.0f + __expf(-v));
                    if (res) v += res[(size_t)row * N + col];
                    if (outf) outf[(size_t)row * N + col] = v;
                    if (outb) outb[(size_t)row * N + col] = f2bf(v);
                }
            }
        }
}

// ------------------------------------------------------------ self-attention
// One block per (b, h, 64-query tile). Scores panel 64x512 resident in LDS.
__global__ __launch_bounds__(256) void self_attn_kernel(
    const unsigned short* __restrict__ qkv, unsigned short* __restrict__ out) {
    extern __shared__ char smem[];
    unsigned short* Ql = (unsigned short*)smem;      // 64x64
    unsigned short* KV = Ql + 64 * 64;               // 128x64 K-chunk / 64x128 V^T-chunk
    float* Sc = (float*)(KV + 128 * 64);             // 64 x 520 (padded)

    const int t = threadIdx.x, lane = t & 31, wid = t >> 5;
    const int qb = blockIdx.x & 7;
    const int bh = blockIdx.x >> 3;
    const int h = bh & 7, b = bh >> 3;
    const size_t rowbase = (size_t)b * NN;
    const int ldq = 3 * INNERD;
    const int m = lane & 15, hi = lane >> 4;
    const int Mt = wid & 3, Ng = wid >> 2;

    {   // stage Q tile 64x64
        int mm = t >> 2, cc = (t & 3) * 16;
        const unsigned short* src =
            qkv + (rowbase + qb * 64 + mm) * ldq + h * DHD + cc;
        *(uint4*)(Ql + mm * 64 + cc)     = *(const uint4*)src;
        *(uint4*)(Ql + mm * 64 + cc + 8) = *(const uint4*)(src + 8);
    }

    // phase 1: S = Q K^T over 4 chunks of 128 keys
    for (int c4 = 0; c4 < 4; ++c4) {
        __syncthreads();
        {   // stage K chunk [128 keys][64 dh]
            int kk = t >> 1, dc = (t & 1) * 32;
            const unsigned short* src =
                qkv + (rowbase + c4 * 128 + kk) * ldq + INNERD + h * DHD + dc;
            #pragma unroll
            for (int j = 0; j < 4; ++j)
                *(uint4*)(KV + kk * 64 + dc + j * 8) = *(const uint4*)(src + j * 8);
        }
        __syncthreads();
        #pragma unroll
        for (int nt = 0; nt < 4; ++nt) {
            int Nt = Ng * 4 + nt;
            v8f acc = {};
            #pragma unroll
            for (int ks = 0; ks < 2; ++ks) {
                Frag a, bk;
                const unsigned short* ap = Ql + (Mt * 16 + m) * 64 + ks * 32;
                a.q[0] = *(const uint4*)(ap + hi * 16);
                a.q[1] = *(const uint4*)(ap + hi * 16 + 8);
                const unsigned short* bp = KV + (Nt * 16 + m) * 64 + ks * 32 + hi * 16;
                bk.q[0] = *(const uint4*)bp; bk.q[1] = *(const uint4*)(bp + 8);
                acc = __builtin_amdgcn_wmma_f32_16x16x32_bf16(
                    false, a.v, false, bk.v, (short)0, acc, false, false);
            }
            #pragma unroll
            for (int j = 0; j < 8; ++j)
                Sc[(Mt * 16 + hi * 8 + j) * 520 + c4 * 128 + Nt * 16 + m] = acc[j];
        }
    }
    __syncthreads();

    // softmax (scale 1/sqrt(64)=0.125): 4 threads per row
    {
        int r = t >> 2, sg = t & 3;
        float* row = Sc + r * 520 + sg * 128;
        float mx = -1e30f;
        for (int j = 0; j < 128; ++j) mx = fmaxf(mx, row[j]);
        mx = fmaxf(mx, __shfl_xor(mx, 1));
        mx = fmaxf(mx, __shfl_xor(mx, 2));
        mx *= 0.125f;
        float sum = 0.f;
        for (int j = 0; j < 128; ++j) {
            float e = __expf(row[j] * 0.125f - mx);
            row[j] = e; sum += e;
        }
        sum += __shfl_xor(sum, 1);
        sum += __shfl_xor(sum, 2);
        float inv = 1.0f / sum;
        for (int j = 0; j < 128; ++j) row[j] *= inv;
    }

    // phase 2: O = P V over 4 chunks of 128 keys
    v8f o0 = {}, o1 = {};
    for (int c4 = 0; c4 < 4; ++c4) {
        __syncthreads();
        {   // stage V^T chunk [64 dh][128 keys]
            int kk = t >> 1, dbase = (t & 1) * 32;
            const unsigned short* src =
                qkv + (rowbase + c4 * 128 + kk) * ldq + 2 * INNERD + h * DHD + dbase;
            #pragma unroll
            for (int j = 0; j < 4; ++j) {
                uint4 v = *(const uint4*)(src + j * 8);
                const unsigned short* pv = (const unsigned short*)&v;
                #pragma unroll
                for (int l = 0; l < 8; ++l)
                    KV[(dbase + j * 8 + l) * 128 + kk] = pv[l];
            }
        }
        __syncthreads();
        #pragma unroll
        for (int ks = 0; ks < 4; ++ks) {
            Frag a, b0, b1;
            const float* prow = Sc + (Mt * 16 + m) * 520 + c4 * 128 + ks * 32;
            #pragma unroll
            for (int i = 0; i < 8; ++i) a.u[i]     = f2bf(prow[hi * 16 + i]);
            #pragma unroll
            for (int i = 0; i < 8; ++i) a.u[8 + i] = f2bf(prow[hi * 16 + 8 + i]);
            const unsigned short* bp0 = KV + ((Ng * 2 + 0) * 16 + m) * 128 + ks * 32 + hi * 16;
            const unsigned short* bp1 = KV + ((Ng * 2 + 1) * 16 + m) * 128 + ks * 32 + hi * 16;
            b0.q[0] = *(const uint4*)bp0; b0.q[1] = *(const uint4*)(bp0 + 8);
            b1.q[0] = *(const uint4*)bp1; b1.q[1] = *(const uint4*)(bp1 + 8);
            o0 = __builtin_amdgcn_wmma_f32_16x16x32_bf16(false, a.v, false, b0.v,
                                                         (short)0, o0, false, false);
            o1 = __builtin_amdgcn_wmma_f32_16x16x32_bf16(false, a.v, false, b1.v,
                                                         (short)0, o1, false, false);
        }
    }
    #pragma unroll
    for (int tile = 0; tile < 2; ++tile) {
        v8f acc; if (tile) acc = o1; else acc = o0;
        int col = h * DHD + (Ng * 2 + tile) * 16 + m;
        #pragma unroll
        for (int j = 0; j < 8; ++j) {
            size_t row = rowbase + qb * 64 + Mt * 16 + hi * 8 + j;
            out[row * INNERD + col] = f2bf(acc[j]);
        }
    }
}

// ----------------------------------------------------------- cross-attention
// One block per (b, h, 64-query tile); 77 keys padded to 96 with masking.
__global__ __launch_bounds__(256) void cross_attn_kernel(
    const unsigned short* __restrict__ qm, const unsigned short* __restrict__ km,
    const unsigned short* __restrict__ vm, unsigned short* __restrict__ out) {
    extern __shared__ char smem[];
    unsigned short* Ql = (unsigned short*)smem;   // 64x64
    unsigned short* Kl = Ql + 64 * 64;            // 96x64
    unsigned short* Vt = Kl + 96 * 64;            // 64x96 (dh-major)
    float* Sc = (float*)(Vt + 64 * 96);           // 64x96

    const int t = threadIdx.x, lane = t & 31, wid = t >> 5;
    const int qblk = blockIdx.x & 7;
    const int bh = blockIdx.x >> 3;
    const int h = bh & 7, b = bh >> 3;
    const int m = lane & 15, hi = lane >> 4;
    const int Mt = wid & 3, Ng = wid >> 2;

    {   // stage Q tile
        int mm = t >> 2, cc = (t & 3) * 16;
        const unsigned short* src =
            qm + ((size_t)b * NN + qblk * 64 + mm) * INNERD + h * DHD + cc;
        *(uint4*)(Ql + mm * 64 + cc)     = *(const uint4*)src;
        *(uint4*)(Ql + mm * 64 + cc + 8) = *(const uint4*)(src + 8);
    }
    for (int idx = t; idx < 96 * 8; idx += 256) {   // stage K (zero-pad >=77)
        int mm = idx >> 3, cc = (idx & 7) * 8;
        uint4 v = make_uint4(0u, 0u, 0u, 0u);
        if (mm < MTOK)
            v = *(const uint4*)(km + ((size_t)b * MTOK + mm) * INNERD + h * DHD + cc);
        *(uint4*)(Kl + mm * 64 + cc) = v;
    }
    for (int idx = t; idx < 96 * 8; idx += 256) {   // stage V^T
        int mm = idx >> 3, cc = (idx & 7) * 8;
        uint4 v = make_uint4(0u, 0u, 0u, 0u);
        if (mm < MTOK)
            v = *(const uint4*)(vm + ((size_t)b * MTOK + mm) * INNERD + h * DHD + cc);
        const unsigned short* pv = (const unsigned short*)&v;
        #pragma unroll
        for (int l = 0; l < 8; ++l) Vt[(cc + l) * 96 + mm] = pv[l];
    }
    __syncthreads();

    // S = Q K^T : 4x6 tiles, wave handles Nt = Ng*3 + {0,1,2}
    #pragma unroll
    for (int nt = 0; nt < 3; ++nt) {
        int Nt = Ng * 3 + nt;
        v8f acc = {};
        #pragma unroll
        for (int ks = 0; ks < 2; ++ks) {
            Frag a, bk;
            const unsigned short* ap = Ql + (Mt * 16 + m) * 64 + ks * 32;
            a.q[0] = *(const uint4*)(ap + hi * 16);
            a.q[1] = *(const uint4*)(ap + hi * 16 + 8);
            const unsigned short* bp = Kl + (Nt * 16 + m) * 64 + ks * 32 + hi * 16;
            bk.q[0] = *(const uint4*)bp; bk.q[1] = *(const uint4*)(bp + 8);
            acc = __builtin_amdgcn_wmma_f32_16x16x32_bf16(
                false, a.v, false, bk.v, (short)0, acc, false, false);
        }
        #pragma unroll
        for (int j = 0; j < 8; ++j)
            Sc[(Mt * 16 + hi * 8 + j) * 96 + Nt * 16 + m] = acc[j];
    }
    __syncthreads();

    {   // masked softmax over 96 cols (valid < 77)
        int r = t >> 2, sg = t & 3;
        float* row = Sc + r * 96 + sg * 24;
        int j0 = sg * 24;
        float mx = -1e30f;
        for (int j = 0; j < 24; ++j)
            if (j0 + j < MTOK) mx = fmaxf(mx, row[j]);
        mx = fmaxf(mx, __shfl_xor(mx, 1));
        mx = fmaxf(mx, __shfl_xor(mx, 2));
        mx *= 0.125f;
        float sum = 0.f;
        for (int j = 0; j < 24; ++j) {
            float e = (j0 + j < MTOK) ? __expf(row[j] * 0.125f - mx) : 0.f;
            row[j] = e; sum += e;
        }
        sum += __shfl_xor(sum, 1);
        sum += __shfl_xor(sum, 2);
        float inv = 1.0f / sum;
        for (int j = 0; j < 24; ++j) row[j] *= inv;
    }
    __syncthreads();

    // O = P V : K-dim 96 -> 3 steps
    v8f o0 = {}, o1 = {};
    #pragma unroll
    for (int ks = 0; ks < 3; ++ks) {
        Frag a, b0, b1;
        const float* prow = Sc + (Mt * 16 + m) * 96 + ks * 32;
        #pragma unroll
        for (int i = 0; i < 8; ++i) a.u[i]     = f2bf(prow[hi * 16 + i]);
        #pragma unroll
        for (int i = 0; i < 8; ++i) a.u[8 + i] = f2bf(prow[hi * 16 + 8 + i]);
        const unsigned short* bp0 = Vt + ((Ng * 2 + 0) * 16 + m) * 96 + ks * 32 + hi * 16;
        const unsigned short* bp1 = Vt + ((Ng * 2 + 1) * 16 + m) * 96 + ks * 32 + hi * 16;
        b0.q[0] = *(const uint4*)bp0; b0.q[1] = *(const uint4*)(bp0 + 8);
        b1.q[0] = *(const uint4*)bp1; b1.q[1] = *(const uint4*)(bp1 + 8);
        o0 = __builtin_amdgcn_wmma_f32_16x16x32_bf16(false, a.v, false, b0.v,
                                                     (short)0, o0, false, false);
        o1 = __builtin_amdgcn_wmma_f32_16x16x32_bf16(false, a.v, false, b1.v,
                                                     (short)0, o1, false, false);
    }
    #pragma unroll
    for (int tile = 0; tile < 2; ++tile) {
        v8f acc; if (tile) acc = o1; else acc = o0;
        int col = h * DHD + (Ng * 2 + tile) * 16 + m;
        #pragma unroll
        for (int j = 0; j < 8; ++j) {
            size_t row = (size_t)b * NN + qblk * 64 + Mt * 16 + hi * 8 + j;
            out[row * INNERD + col] = f2bf(acc[j]);
        }
    }
}

// ------------------------------------------------------------------- driver
extern "C" void kernel_launch(void* const* d_in, const int* in_sizes, int n_in,
                              void* d_out, int out_size, void* d_ws, size_t ws_size,
                              hipStream_t stream) {
    (void)in_sizes; (void)n_in; (void)out_size; (void)ws_size;
    const float* x     = (const float*)d_in[0];
    const float* cond  = (const float*)d_in[1];
    const float* ln1_g = (const float*)d_in[2];
    const float* ln1_b = (const float*)d_in[3];
    const float* w_qkv = (const float*)d_in[4];
    const float* w_so  = (const float*)d_in[5];
    const float* b_so  = (const float*)d_in[6];
    const float* ln2_g = (const float*)d_in[7];
    const float* ln2_b = (const float*)d_in[8];
    const float* w_cq  = (const float*)d_in[9];
    const float* w_ck  = (const float*)d_in[10];
    const float* w_cv  = (const float*)d_in[11];
    const float* w_co  = (const float*)d_in[12];
    const float* b_co  = (const float*)d_in[13];
    const float* ln3_g = (const float*)d_in[14];
    const float* ln3_b = (const float*)d_in[15];
    const float* w_m1  = (const float*)d_in[16];
    const float* b_m1  = (const float*)d_in[17];
    const float* w_m2  = (const float*)d_in[18];
    const float* b_m2  = (const float*)d_in[19];

    char* ws = (char*)d_ws;
    size_t off = 0;
    auto alloc = [&](size_t bytes) -> char* {
        char* p = ws + off;
        off = (off + bytes + 255) & ~(size_t)255;
        return p;
    };
    float* xt             = (float*)alloc(4096ull * 512 * 4);
    float* yy             = (float*)alloc(4096ull * 512 * 4);
    unsigned short* ybf   = (unsigned short*)alloc(4096ull * 512 * 2);
    unsigned short* qkvbf = (unsigned short*)alloc(4096ull * 1536 * 2);
    unsigned short* qbf   = (unsigned short*)alloc(4096ull * 512 * 2);
    unsigned short* ckbf  = (unsigned short*)alloc(616ull * 512 * 2);
    unsigned short* cvbf  = (unsigned short*)alloc(616ull * 512 * 2);
    unsigned short* aobf  = (unsigned short*)alloc(4096ull * 512 * 2);
    unsigned short* hbf   = (unsigned short*)alloc(4096ull * 2048 * 2);
    unsigned short* condbf = (unsigned short*)alloc(473088ull * 2);
    // weights, pre-transposed per layer: Wt [N][K] bf16
    unsigned short* wqkvbf = (unsigned short*)alloc(3145728ull * 2);
    unsigned short* wsobf  = (unsigned short*)alloc(1048576ull * 2);
    unsigned short* wcqbf  = (unsigned short*)alloc(1048576ull * 2);
    unsigned short* wckbf  = (unsigned short*)alloc(1572864ull * 2);
    unsigned short* wcvbf  = (unsigned short*)alloc(1572864ull * 2);
    unsigned short* wcobf  = (unsigned short*)alloc(1048576ull * 2);
    unsigned short* wm1bf  = (unsigned short*)alloc(4194304ull * 2);
    unsigned short* wm2bf  = (unsigned short*)alloc(4194304ull * 2);

    cvt_bf16_kernel<<<(473088 + 255) / 256, 256, 0, stream>>>(cond, condbf, 473088);
    auto cvtw = [&](const float* s, unsigned short* d, int K, int N) {
        int total = DEPTHN * K * N;
        cvt_wt_kernel<<<(total + 255) / 256, 256, 0, stream>>>(s, d, K, N, total);
    };
    cvtw(w_qkv, wqkvbf, 512, 1536);
    cvtw(w_so,  wsobf,  512, 512);
    cvtw(w_cq,  wcqbf,  512, 512);
    cvtw(w_ck,  wckbf,  768, 512);
    cvtw(w_cv,  wcvbf,  768, 512);
    cvtw(w_co,  wcobf,  512, 512);
    cvtw(w_m1,  wm1bf,  512, 2048);
    cvtw(w_m2,  wm2bf,  2048, 512);

    transpose_in_kernel<<<8192, 256, 0, stream>>>(x, xt);

    auto gemm = [&](const unsigned short* A, const unsigned short* Wt,
                    const float* bias, const float* res, float* outf,
                    unsigned short* outb, int R, int K, int N, int silu) {
        dim3 grid((R + 127) / 128, N / 128);
        gemm_bf16_kernel<<<grid, 256, 0, stream>>>(A, Wt, bias, res, outf, outb,
                                                   R, K, N, silu);
    };

    const size_t smem_self  = 64 * 64 * 2 + 128 * 64 * 2 + 64 * 520 * 4;
    const size_t smem_cross = 64 * 64 * 2 + 96 * 64 * 2 + 64 * 96 * 2 + 64 * 96 * 4;

    for (int i = 0; i < DEPTHN; ++i) {
        // self-attention block
        ln_kernel<<<4096, 128, 0, stream>>>(xt, ln1_g + i * 512, ln1_b + i * 512, yy, ybf);
        gemm(ybf, wqkvbf + (size_t)i * 786432, nullptr, nullptr, nullptr, qkvbf,
             4096, 512, 1536, 0);
        self_attn_kernel<<<512, 256, smem_self, stream>>>(qkvbf, aobf);
        gemm(aobf, wsobf + (size_t)i * 262144, b_so + i * 512, yy, xt, nullptr,
             4096, 512, 512, 0);
        // cross-attention block
        ln_kernel<<<4096, 128, 0, stream>>>(xt, ln2_g + i * 512, ln2_b + i * 512, yy, ybf);
        gemm(ybf, wcqbf + (size_t)i * 262144, nullptr, nullptr, nullptr, qbf,
             4096, 512, 512, 0);
        gemm(condbf, wckbf + (size_t)i * 393216, nullptr, nullptr, nullptr, ckbf,
             616, 768, 512, 0);
        gemm(condbf, wcvbf + (size_t)i * 393216, nullptr, nullptr, nullptr, cvbf,
             616, 768, 512, 0);
        cross_attn_kernel<<<512, 256, smem_cross, stream>>>(qbf, ckbf, cvbf, aobf);
        gemm(aobf, wcobf + (size_t)i * 262144, b_co + i * 512, yy, xt, nullptr,
             4096, 512, 512, 0);
        // SiLU MLP block
        ln_kernel<<<4096, 128, 0, stream>>>(xt, ln3_g + i * 512, ln3_b + i * 512, yy, ybf);
        gemm(ybf, wm1bf + (size_t)i * 1048576, b_m1 + i * 2048, nullptr, nullptr,
             hbf, 4096, 512, 2048, 1);
        gemm(hbf, wm2bf + (size_t)i * 1048576, b_m2 + i * 512, yy, xt, nullptr,
             4096, 2048, 512, 0);
    }
    transpose_out_kernel<<<8192, 256, 0, stream>>>(xt, (float*)d_out);
}
